// CustomLSTMCell_25366076850664
// MI455X (gfx1250) — matrix operand order repaired and checked by reading, not compiled, not run on hardware
//
#include <hip/hip_runtime.h>

#define D_ 512
#define F_ 1024
#define M_TOTAL 16384          // B*P rows
#define K_TOTAL (F_ + D_)      // 1536 concatenated reduction dim
#define KT 32                  // K per WMMA step (bf16 16x16x32)
#define NK (K_TOTAL / KT)      // 48 K-steps
#define BM 128                 // block tile M
#define BN 64                  // block tile N
#define LDSS (KT + 8)          // padded LDS row stride (40 shorts = 80B, 16B-mult)

typedef __attribute__((ext_vector_type(16))) __bf16       v16bf;
typedef __attribute__((ext_vector_type(8)))  __bf16       v8bf;
typedef __attribute__((ext_vector_type(8)))  float        v8f;
typedef __attribute__((ext_vector_type(4)))  unsigned int u32x4;
typedef __attribute__((ext_vector_type(4)))  float        f32x4;

union FragBF { v16bf v; u32x4 u[2]; };

struct Params {
    const float* z;
    const float* c_prev;
    const float* m_prev;
    const float* h_prev;
    const float* n_prev;
    const float* mask;
    const float* W[4];   // (F,F)  gates i,f,o,z
    const float* Wb[4];  // (F,)
    const float* R[4];   // (F,D)
    const float* Rb[4];  // (F,)
    float* out_c;
    float* out_m;
    float* out_h;
    float* out_n;
};

// 8 fp32 -> 8 bf16 via native converts (v_cvt_*bf16*, co-executes with WMMA)
__device__ __forceinline__ u32x4 cvt8(f32x4 f0, f32x4 f1) {
    v8bf b;
    b[0] = (__bf16)f0.x; b[1] = (__bf16)f0.y; b[2] = (__bf16)f0.z; b[3] = (__bf16)f0.w;
    b[4] = (__bf16)f1.x; b[5] = (__bf16)f1.y; b[6] = (__bf16)f1.z; b[7] = (__bf16)f1.w;
    return __builtin_bit_cast(u32x4, b);
}

// registers holding one block tile's worth of next-K global data per thread
struct StageRegs {
    f32x4 a[4];      // 16 floats: A tile, one row segment
    f32x4 w[4][2];   // 8 floats per gate: weight tiles
};

__device__ __forceinline__ void stage_load(const Params& p, int k0,
                                           int block_m0, int block_n0, int tid,
                                           StageRegs& r) {
    // A tile: 128 rows x 32 cols; thread -> row tid>>1, 16 cols at (tid&1)*16
    const int ar = tid >> 1;
    const int ac = (tid & 1) * 16;
    const int mrow = block_m0 + ar;
    const float* asrc = (k0 < F_)
        ? p.h_prev + (size_t)mrow * F_ + (k0 + ac)
        : p.z      + (size_t)mrow * D_ + (k0 - F_ + ac);
    r.a[0] = ((const f32x4*)asrc)[0];
    r.a[1] = ((const f32x4*)asrc)[1];
    r.a[2] = ((const f32x4*)asrc)[2];
    r.a[3] = ((const f32x4*)asrc)[3];
    // W tiles: 4 x (64 rows x 32 cols); thread -> row tid>>2, 8 cols at (tid&3)*8
    const int wr = tid >> 2;
    const int wc = (tid & 3) * 8;
    const int nrow = block_n0 + wr;
    #pragma unroll
    for (int g = 0; g < 4; ++g) {
        const float* wsrc = (k0 < F_)
            ? p.W[g] + (size_t)nrow * F_ + (k0 + wc)
            : p.R[g] + (size_t)nrow * D_ + (k0 - F_ + wc);
        r.w[g][0] = ((const f32x4*)wsrc)[0];
        r.w[g][1] = ((const f32x4*)wsrc)[1];
    }
}

__device__ __forceinline__ void stage_store(unsigned short (*sA)[LDSS],
                                            unsigned short (*sW)[BN][LDSS],
                                            int tid, const StageRegs& r) {
    const int ar = tid >> 1;
    const int ac = (tid & 1) * 16;
    *(u32x4*)&sA[ar][ac]     = cvt8(r.a[0], r.a[1]);
    *(u32x4*)&sA[ar][ac + 8] = cvt8(r.a[2], r.a[3]);
    const int wr = tid >> 2;
    const int wc = (tid & 3) * 8;
    #pragma unroll
    for (int g = 0; g < 4; ++g)
        *(u32x4*)&sW[g][wr][wc] = cvt8(r.w[g][0], r.w[g][1]);
}

// 16 WMMAs: 4 M-subtiles x 4 gates; each B fragment feeds 4 back-to-back WMMAs
__device__ __forceinline__ void compute_step(const unsigned short (*sA)[LDSS],
                                             const unsigned short (*sW)[BN][LDSS],
                                             int wm, int wn, int lane,
                                             v8f acc[4][4]) {
    const int fm  = lane & 15;
    const int fhi = lane >> 4;
    const int a_kb = fhi * 8;    // ISA 16-bit A 16x32 layout
    const int b_kb = fhi * 8;    // same k order as A (8h+s, 16+8h+s-8)
    FragBF af[4];
    #pragma unroll
    for (int m = 0; m < 4; ++m) {
        const int arow = wm * 64 + m * 16 + fm;
        af[m].u[0] = *(const u32x4*)&sA[arow][a_kb];
        af[m].u[1] = *(const u32x4*)&sA[arow][a_kb + 16];
    }
    const int brow = wn * 16 + fm;
    #pragma unroll
    for (int g = 0; g < 4; ++g) {
        FragBF bf_;
        bf_.u[0] = *(const u32x4*)&sW[g][brow][b_kb];
        bf_.u[1] = *(const u32x4*)&sW[g][brow][b_kb + 16];
        #pragma unroll
        for (int m = 0; m < 4; ++m)
            acc[g][m] = __builtin_amdgcn_wmma_f32_16x16x32_bf16(
                false, af[m].v, false, bf_.v, (short)0, acc[g][m],
                false, false);
    }
}

__launch_bounds__(256, 1)
__global__ void mlstm_fused_wmma(Params p) {
    __shared__ __align__(16) unsigned short sA[2][BM][LDSS];      // 2 x 10 KB
    __shared__ __align__(16) unsigned short sW[2][4][BN][LDSS];   // 2 x 20 KB

    const int tid  = threadIdx.x;
    const int lane = tid & 31;
    const int wave = tid >> 5;
    const int wm   = wave & 1;   // 2 M-waves x 64 rows
    const int wn   = wave >> 1;  // 4 N-waves x 16 cols

    const int block_n0 = blockIdx.x * BN;
    const int block_m0 = blockIdx.y * BM;

    v8f acc[4][4];   // [gate][m-subtile]
    #pragma unroll
    for (int g = 0; g < 4; ++g)
        #pragma unroll
        for (int m = 0; m < 4; ++m)
            #pragma unroll
            for (int j = 0; j < 8; ++j)
                acc[g][m][j] = 0.0f;

    StageRegs r;
    stage_load(p, 0, block_m0, block_n0, tid, r);
    stage_store(sA[0], sW[0], tid, r);
    __syncthreads();

    for (int kk = 0; kk < NK; ++kk) {
        const int cur = kk & 1;
        if (kk + 1 < NK)  // issue next tile's global loads before the WMMAs
            stage_load(p, (kk + 1) * KT, block_m0, block_n0, tid, r);
        compute_step(sA[cur], sW[cur], wm, wn, lane, acc);
        if (kk + 1 < NK)  // convert + write other buffer after compute
            stage_store(sA[cur ^ 1], sW[cur ^ 1], tid, r);
        __syncthreads();
    }

    // ---- epilogue: biases + max-stabilized exponential gating ----
    // C/D layout: element v -> row v + 8*(lane>>4), col = lane&15
    const int fm  = lane & 15;
    const int fhi = lane >> 4;
    const int ncol = block_n0 + wn * 16 + fm;   // one column per lane
    const float bi_ = p.Wb[0][ncol] + p.Rb[0][ncol];
    const float bf_ = p.Wb[1][ncol] + p.Rb[1][ncol];
    const float bo_ = p.Wb[2][ncol] + p.Rb[2][ncol];
    const float bz_ = p.Wb[3][ncol] + p.Rb[3][ncol];
    #pragma unroll
    for (int m = 0; m < 4; ++m) {
        const int rbase = block_m0 + wm * 64 + m * 16 + fhi * 8;
        #pragma unroll
        for (int v = 0; v < 8; ++v) {
            const int row = rbase + v;
            const size_t idx = (size_t)row * F_ + ncol;
            const float it = acc[0][m][v] + bi_;
            const float ft = acc[1][m][v] + bf_;
            const float ot = acc[2][m][v] + bo_;
            const float zt = acc[3][m][v] + bz_;
            const float mp  = p.m_prev[idx];
            const float msk = p.mask[row];
            const float m_t = fmaxf(ft + mp, it);
            const float i_g = __expf(it - m_t);
            const float f_g = __expf(ft + mp - m_t);
            const float o_g = 1.0f / (1.0f + __expf(-ot));
            const float n_t = f_g * p.n_prev[idx] + i_g;
            const float z_g = tanhf(zt);
            const float cp  = p.c_prev[idx];
            const float c_new = cp * f_g + z_g * i_g;
            const float c_t = msk * c_new + (1.0f - msk) * cp;
            const float h_t = msk * (o_g * (c_t / n_t)) + (1.0f - msk) * p.h_prev[idx];
            p.out_c[idx] = c_t;
            p.out_m[idx] = m_t;
            p.out_h[idx] = h_t;
            p.out_n[idx] = n_t;
        }
    }
}

extern "C" void kernel_launch(void* const* d_in, const int* in_sizes, int n_in,
                              void* d_out, int out_size, void* d_ws, size_t ws_size,
                              hipStream_t stream) {
    (void)in_sizes; (void)n_in; (void)d_ws; (void)ws_size; (void)out_size;
    Params p;
    p.z      = (const float*)d_in[0];
    p.c_prev = (const float*)d_in[1];
    p.m_prev = (const float*)d_in[2];
    p.h_prev = (const float*)d_in[3];
    p.n_prev = (const float*)d_in[4];
    p.mask   = (const float*)d_in[5];
    p.W[0] = (const float*)d_in[6];  p.Wb[0] = (const float*)d_in[7];
    p.W[1] = (const float*)d_in[8];  p.Wb[1] = (const float*)d_in[9];
    p.W[2] = (const float*)d_in[10]; p.Wb[2] = (const float*)d_in[11];
    p.W[3] = (const float*)d_in[12]; p.Wb[3] = (const float*)d_in[13];
    p.R[0] = (const float*)d_in[14]; p.Rb[0] = (const float*)d_in[15];
    p.R[1] = (const float*)d_in[16]; p.Rb[1] = (const float*)d_in[17];
    p.R[2] = (const float*)d_in[18]; p.Rb[2] = (const float*)d_in[19];
    p.R[3] = (const float*)d_in[20]; p.Rb[3] = (const float*)d_in[21];
    const size_t MF = (size_t)M_TOTAL * F_;
    float* out = (float*)d_out;
    p.out_c = out;
    p.out_m = out + MF;
    p.out_h = out + 2 * MF;
    p.out_n = out + 3 * MF;
    dim3 grid(F_ / BN, M_TOTAL / BM);   // (16, 128) = 2048 workgroups
    dim3 block(256);                     // 8 wave32s
    hipLaunchKernelGGL(mlstm_fused_wmma, grid, block, 0, stream, p);
}
